// PairwiseCosineSimilarity_22780506538335
// MI455X (gfx1250) — matrix operand order repaired and checked by reading, not compiled, not run on hardware
//
#include <hip/hip_runtime.h>

// Problem constants (match the reference).
#define B_   16
#define NX_  2048
#define NY_  2048
#define L_   256
#define EPS_ 1e-8f

typedef __attribute__((ext_vector_type(16))) __bf16 v16bf;
typedef __attribute__((ext_vector_type(8)))  __bf16 v8bf;
typedef __attribute__((ext_vector_type(8)))  float  v8f;
typedef __attribute__((ext_vector_type(4)))  float  v4f;

// -----------------------------------------------------------------------------
// Kernel 1: fp32 row -> split bf16 (hi + lo) planes, plus fp32 L2 norm per row.
// One wave (32 lanes) per row of 256 elements; each lane handles 8 contiguous.
// -----------------------------------------------------------------------------
__global__ __launch_bounds__(256) void cvt_norm_kernel(
    const float* __restrict__ src,   // [rows, L_]
    __bf16* __restrict__ hi,         // [rows, L_]
    __bf16* __restrict__ lo,         // [rows, L_]
    float*  __restrict__ nrm,        // [rows]
    int rows)
{
    const int wave = threadIdx.x >> 5;
    const int lane = threadIdx.x & 31;
    const int row  = blockIdx.x * 8 + wave;
    if (row >= rows) return;

    const float* r = src + (size_t)row * L_ + lane * 8;
    v4f a = *(const v4f*)(r);
    v4f b = *(const v4f*)(r + 4);

    float ss = 0.f;
    v8bf h, l;
#pragma unroll
    for (int e = 0; e < 4; ++e) {
        float x = a[e];
        ss += x * x;
        __bf16 xh = (__bf16)x;          // RNE truncation: hi part
        h[e] = xh;
        l[e] = (__bf16)(x - (float)xh); // residual: lo part
    }
#pragma unroll
    for (int e = 0; e < 4; ++e) {
        float x = b[e];
        ss += x * x;
        __bf16 xh = (__bf16)x;
        h[4 + e] = xh;
        l[4 + e] = (__bf16)(x - (float)xh);
    }

    *(v8bf*)(hi + (size_t)row * L_ + lane * 8) = h;
    *(v8bf*)(lo + (size_t)row * L_ + lane * 8) = l;

    // wave32 reduction of sum of squares
#pragma unroll
    for (int off = 16; off > 0; off >>= 1)
        ss += __shfl_xor(ss, off, 32);
    if (lane == 0) nrm[row] = sqrtf(ss);
}

// -----------------------------------------------------------------------------
// Kernel 2: batched GEMM via V_WMMA_F32_16X16X32_BF16 with split-bf16 (3 WMMA
// per tile pair: hi*hi + hi*lo + lo*hi), fused cosine epilogue.
//
// Block = 256 threads = 8 waves; block tile 128(M) x 128(N); wave grid 2x4,
// each wave computes 64x32 = 4x2 subtiles of 16x16. K loop: 8 steps of 32.
//
// Fragments are loaded directly in the CDNA5 WMMA VGPR layout:
//   A (16x32 bf16, MxK): lane m = M (l16), half = lane>>4
//       elems 0..7  = K = kk + 8*half + (0..7)
//       elems 8..15 = K = kk + 16 + 8*half + (0..7)
//   B (32x16 bf16, KxN): lane n = N (l16), half = lane>>4
//       elems 0..7  = K = kk + 8*half + (0..7)
//       elems 8..15 = K = kk + 16 + 8*half + (0..7)
//   C/D (16x16 f32): elem r -> M = r + 8*half, N = l16
// -----------------------------------------------------------------------------
__global__ __launch_bounds__(256) void cosgemm_kernel(
    const __bf16* __restrict__ Xh, const __bf16* __restrict__ Xl,
    const __bf16* __restrict__ Yh, const __bf16* __restrict__ Yl,
    const float*  __restrict__ nx, const float*  __restrict__ ny,
    float* __restrict__ out)
{
    const int b     = blockIdx.z;
    const int lane  = threadIdx.x & 31;
    const int w     = threadIdx.x >> 5;
    const int waveM = w & 1;         // 0..1
    const int waveN = w >> 1;        // 0..3
    const int half  = lane >> 4;
    const int l16   = lane & 15;

    const int m0 = blockIdx.x * 128 + waveM * 64;  // wave M origin
    const int n0 = blockIdx.y * 128 + waveN * 32;  // wave N origin

    v8f acc[4][2];
#pragma unroll
    for (int i = 0; i < 4; ++i)
#pragma unroll
        for (int j = 0; j < 2; ++j)
            acc[i][j] = (v8f){0.f, 0.f, 0.f, 0.f, 0.f, 0.f, 0.f, 0.f};

    // per-lane row bases (element index into [B, N, L] bf16 planes)
    const size_t xrow = ((size_t)b * NX_ + (size_t)(m0 + l16)) * L_;
    const size_t yrow = ((size_t)b * NY_ + (size_t)(n0 + l16)) * L_;

    for (int kk = 0; kk < L_; kk += 32) {
        v16bf Ah[4], Al[4], Bh[2], Bl[2];

#pragma unroll
        for (int i = 0; i < 4; ++i) {
            const size_t base = xrow + (size_t)i * 16 * L_ + kk + 8 * half;
            v8bf h0 = *(const v8bf*)(Xh + base);
            v8bf h1 = *(const v8bf*)(Xh + base + 16);
            Ah[i] = __builtin_shufflevector(h0, h1,
                    0,1,2,3,4,5,6,7,8,9,10,11,12,13,14,15);
            v8bf l0 = *(const v8bf*)(Xl + base);
            v8bf l1 = *(const v8bf*)(Xl + base + 16);
            Al[i] = __builtin_shufflevector(l0, l1,
                    0,1,2,3,4,5,6,7,8,9,10,11,12,13,14,15);
        }
#pragma unroll
        for (int j = 0; j < 2; ++j) {
            const size_t base = yrow + (size_t)j * 16 * L_ + kk + 8 * half;
            v8bf h0 = *(const v8bf*)(Yh + base);
            v8bf h1 = *(const v8bf*)(Yh + base + 16);
            Bh[j] = __builtin_shufflevector(h0, h1,
                    0,1,2,3,4,5,6,7,8,9,10,11,12,13,14,15);
            v8bf l0 = *(const v8bf*)(Yl + base);
            v8bf l1 = *(const v8bf*)(Yl + base + 16);
            Bl[j] = __builtin_shufflevector(l0, l1,
                    0,1,2,3,4,5,6,7,8,9,10,11,12,13,14,15);
        }

#pragma unroll
        for (int i = 0; i < 4; ++i)
#pragma unroll
            for (int j = 0; j < 2; ++j) {
                acc[i][j] = __builtin_amdgcn_wmma_f32_16x16x32_bf16(
                    false, Ah[i], false, Bh[j], (short)0, acc[i][j], false, false);
                acc[i][j] = __builtin_amdgcn_wmma_f32_16x16x32_bf16(
                    false, Ah[i], false, Bl[j], (short)0, acc[i][j], false, false);
                acc[i][j] = __builtin_amdgcn_wmma_f32_16x16x32_bf16(
                    false, Al[i], false, Bh[j], (short)0, acc[i][j], false, false);
            }
    }

    // ---------------- epilogue: cosine normalization + store ----------------
    float nyv[2];
#pragma unroll
    for (int j = 0; j < 2; ++j)
        nyv[j] = ny[(size_t)b * NY_ + n0 + j * 16 + l16];

#pragma unroll
    for (int i = 0; i < 4; ++i) {
        const float* np = nx + (size_t)b * NX_ + m0 + i * 16 + 8 * half;
        v4f nA = *(const v4f*)(np);
        v4f nB = *(const v4f*)(np + 4);
#pragma unroll
        for (int j = 0; j < 2; ++j) {
            const int ncol = n0 + j * 16 + l16;
#pragma unroll
            for (int r = 0; r < 8; ++r) {
                const float nxv   = (r < 4) ? nA[r] : nB[r - 4];
                const float denom = fmaxf(nxv * nyv[j], EPS_);
                const size_t mrow = (size_t)b * NX_ + m0 + i * 16 + 8 * half + r;
                out[mrow * NY_ + ncol] = acc[i][j][r] / denom;
            }
        }
    }
}

// -----------------------------------------------------------------------------
// Launch. Workspace layout (needs ~64.25 MB):
//   Xh (16 MB) | Xl (16 MB) | Yh (16 MB) | Yl (16 MB) | nx (128 KB) | ny (128 KB)
// -----------------------------------------------------------------------------
extern "C" void kernel_launch(void* const* d_in, const int* in_sizes, int n_in,
                              void* d_out, int out_size, void* d_ws, size_t ws_size,
                              hipStream_t stream) {
    (void)in_sizes; (void)n_in; (void)out_size; (void)ws_size;

    const float* x = (const float*)d_in[0];   // [B, NX, L]
    const float* y = (const float*)d_in[1];   // [B, NY, L]
    float* out = (float*)d_out;               // [B, NX, NY]

    const size_t nXe = (size_t)B_ * NX_ * L_; // 8,388,608
    const size_t nYe = (size_t)B_ * NY_ * L_;

    __bf16* Xh = (__bf16*)d_ws;
    __bf16* Xl = Xh + nXe;
    __bf16* Yh = Xl + nXe;
    __bf16* Yl = Yh + nYe;
    float*  nx = (float*)(Yl + nYe);
    float*  ny = nx + (size_t)B_ * NX_;

    // Phase 1: convert + norms (wave-per-row; 8 rows per 256-thread block)
    cvt_norm_kernel<<<(B_ * NX_) / 8, 256, 0, stream>>>(x, Xh, Xl, nx, B_ * NX_);
    cvt_norm_kernel<<<(B_ * NY_) / 8, 256, 0, stream>>>(y, Yh, Yl, ny, B_ * NY_);

    // Phase 2: WMMA GEMM + fused cosine epilogue
    dim3 grid(NX_ / 128, NY_ / 128, B_);
    cosgemm_kernel<<<grid, 256, 0, stream>>>(Xh, Xl, Yh, Yl, nx, ny, out);
}
